// GatedLinearMemory_22780506538741
// MI455X (gfx1250) — compile-verified, matrix operand order checked by reading, not run on hardware
//
#include <hip/hip_runtime.h>
#include <hip/hip_bf16.h>

// Problem dims (fixed by reference)
#define BB 8
#define LL 4096
#define DD 1024
#define MM 64
#define CC 64                 // scan chunk length (== M)
#define NROWS (BB * LL)       // 32768 rows for projection / output GEMMs
#define DECAY_MIN 0.9f
#define DECAY_MAX 0.999f

typedef __attribute__((ext_vector_type(16))) __bf16 v16bf;
typedef __attribute__((ext_vector_type(8)))  __bf16 v8bf;
typedef __attribute__((ext_vector_type(8)))  float  v8f;

#define WMMA_BF16(A, B, C) \
    __builtin_amdgcn_wmma_f32_16x16x32_bf16(false, (A), false, (B), (short)0, (C), false, false)

// ---- CDNA5 async global->LDS copies (ASYNCcnt), with safe fallback --------
#if defined(__gfx1250__) && __has_builtin(__builtin_amdgcn_global_load_async_to_lds_b32) && \
    __has_builtin(__builtin_amdgcn_s_wait_asynccnt)
#define HAVE_ASYNC_LDS 1
#else
#define HAVE_ASYNC_LDS 0
#endif

__device__ __forceinline__ void cp_async_b32(const void* g, void* l) {
#if HAVE_ASYNC_LDS
    __builtin_amdgcn_global_load_async_to_lds_b32(
        (__attribute__((address_space(1))) int*)g,
        (__attribute__((address_space(3))) int*)l, 0, 0);
#else
    *(unsigned int*)l = *(const unsigned int*)g;
#endif
}
__device__ __forceinline__ void cp_async_wait() {
#if HAVE_ASYNC_LDS
    __builtin_amdgcn_s_wait_asynccnt(0);
#endif
}

// ---- WMMA 16-bit fragment geometry (ISA 7.12.2) ---------------------------
// lane = half*16 + (d&15); element i of the lane maps to k = base(i) + 8*half
__device__ __forceinline__ int wmma_kk(int i /*0..15*/, int half) {
    int r = i >> 1;
    int base = (r < 4) ? (2 * r) : (16 + 2 * (r - 4));
    return base + 8 * half + (i & 1);
}

// Fragment-major LDS slot of element (d16 = row-for-A / col-for-B, k).
// ntiles = number of 16-wide tiles along d16. Lane's 16 elems are contiguous.
__device__ __forceinline__ int fslot(int d16, int k, int ntiles) {
    int f    = (k >> 5) * ntiles + (d16 >> 4);
    int kk   = k & 31;
    int half = (kk >> 3) & 1;
    int i    = (kk & 7) + ((kk >> 4) << 3);
    return (((f << 5) + (half << 4) + (d16 & 15)) << 4) + i;
}

// Load one 32-byte fragment (fragIdx = (k0>>5)*ntiles + tile16)
__device__ __forceinline__ v16bf frag_ld(const __bf16* buf, int fragIdx, int lane) {
    return *(const v16bf*)(buf + (((fragIdx << 5) + lane) << 4));
}

__device__ __forceinline__ float sigmoidf_(float x) { return 1.0f / (1.0f + __expf(-x)); }

// ---------------------------------------------------------------------------
// Kernel 1: gates[n] = sigmoid(x[n,:] . Wg + bg)
__global__ __launch_bounds__(256) void gate_kernel(const float* __restrict__ x,
                                                   const float* __restrict__ Wg,
                                                   const float* __restrict__ bg,
                                                   float* __restrict__ gates) {
    int n = blockIdx.x * blockDim.x + threadIdx.x;
    if (n >= NROWS) return;
    const float* xr = x + (size_t)n * DD;
    float acc = 0.f;
    for (int d = 0; d < DD; d += 4) {
        float4 xv = *(const float4*)(xr + d);
        float4 wv = *(const float4*)(Wg + d);
        acc += xv.x * wv.x + xv.y * wv.y + xv.z * wv.z + xv.w * wv.w;
    }
    gates[n] = sigmoidf_(acc + bg[0]);
}

// ---------------------------------------------------------------------------
// Kernel 2: fused q/k/v projection with LDS tiling.
// Block = 128 rows x 192 cols; x staged once per 32-k slice (bf16 pairs).
// Epilogue applies gates and emits: q (normal), g*k (normal + transposed),
// g*v (transposed), so scan staging is pure-copy (async-friendly).
__global__ __launch_bounds__(256) void proj_kernel(const float* __restrict__ x,
                                                   const float* __restrict__ Wq,
                                                   const float* __restrict__ Wk,
                                                   const float* __restrict__ Wv,
                                                   const float* __restrict__ gates,
                                                   __bf16* __restrict__ qb,
                                                   __bf16* __restrict__ kgb,
                                                   __bf16* __restrict__ ktb,
                                                   __bf16* __restrict__ vtb) {
    __shared__ __align__(32) __bf16 Xs[128 * 32];   // A-layout, 8 frags
    __shared__ __align__(32) __bf16 Ws[192 * 32];   // B-layout, 12 frags

    const int tid = threadIdx.x, lane = tid & 31, wave = tid >> 5;
    const int row0 = blockIdx.x * 128;

    v8f zeroV = {};
    v8f acc[12];
#pragma unroll
    for (int j = 0; j < 12; ++j) acc[j] = zeroV;

    for (int ks = 0; ks < DD; ks += 32) {
        for (int p = tid; p < 2048; p += 256) {           // X: 128 rows x 16 k-pairs
            int r = p >> 4, k2 = (p & 15) << 1;
            float2 xv = *(const float2*)(x + (size_t)(row0 + r) * DD + ks + k2);
            union { unsigned int u; __bf16 h[2]; } pk;
            pk.h[0] = (__bf16)xv.x; pk.h[1] = (__bf16)xv.y;
            *(unsigned int*)&Xs[fslot(r, k2, 8)] = pk.u;
        }
        for (int p = tid; p < 3072; p += 256) {           // W: 192 cols x 16 k-pairs
            int c = p >> 4, k2 = (p & 15) << 1;
            const float* W = (c < 64) ? Wq : (c < 128) ? Wk : Wv;
            int c64 = c & 63;
            float w0 = W[(size_t)(ks + k2) * MM + c64];
            float w1 = W[(size_t)(ks + k2 + 1) * MM + c64];
            union { unsigned int u; __bf16 h[2]; } pk;
            pk.h[0] = (__bf16)w0; pk.h[1] = (__bf16)w1;
            *(unsigned int*)&Ws[fslot(c, k2, 12)] = pk.u;
        }
        __syncthreads();
        v16bf a = frag_ld(Xs, wave, lane);                // one A frag, reused 12x
#pragma unroll
        for (int j = 0; j < 12; ++j) {
            v16bf bf = frag_ld(Ws, j, lane);
            acc[j] = WMMA_BF16(a, bf, acc[j]);
        }
        __syncthreads();
    }

    // ---- epilogue ----------------------------------------------------------
    const int half = lane >> 4, n15 = lane & 15;
    const int mrow0 = row0 + wave * 16;
    float gv[8];
#pragma unroll
    for (int r = 0; r < 8; ++r) gv[r] = gates[mrow0 + r + 8 * half];
    const int bidx = mrow0 >> 12;                 // batch index
    const int l0 = (mrow0 & 4095) + 8 * half;     // seq position of acc row 0

#pragma unroll
    for (int j = 0; j < 4; ++j) {                 // q tiles -> normal layout
        int nc = j * 16 + n15;
#pragma unroll
        for (int r = 0; r < 8; ++r)
            qb[(size_t)(mrow0 + r + 8 * half) * MM + nc] = (__bf16)acc[j][r];
    }
#pragma unroll
    for (int j = 0; j < 4; ++j) {                 // k tiles -> g*k normal + transposed
        int nc = j * 16 + n15;
        v8bf pk;
#pragma unroll
        for (int r = 0; r < 8; ++r) {
            float kv = acc[4 + j][r] * gv[r];
            kgb[(size_t)(mrow0 + r + 8 * half) * MM + nc] = (__bf16)kv;
            pk[r] = (__bf16)kv;
        }
        *(v8bf*)(ktb + (size_t)(bidx * MM + nc) * LL + l0) = pk;   // 16B store
    }
#pragma unroll
    for (int j = 0; j < 4; ++j) {                 // v tiles -> g*v transposed only
        int nc = j * 16 + n15;
        v8bf pk;
#pragma unroll
        for (int r = 0; r < 8; ++r) pk[r] = (__bf16)(acc[8 + j][r] * gv[r]);
        *(v8bf*)(vtb + (size_t)(bidx * MM + nc) * LL + l0) = pk;
    }
}

// ---------------------------------------------------------------------------
// Kernel 3: chunked gated fast-weight scan; 1 workgroup (8 waves) per batch.
// State S lives in registers (2 tiles per wave); bf16 copy in LDS for WMMA.
// All chunk staging is async global->LDS pair copies into fragment layout.
__global__ __launch_bounds__(256) void scan_kernel(const __bf16* __restrict__ qb,
                                                   const __bf16* __restrict__ kgb,
                                                   const __bf16* __restrict__ ktb,
                                                   const __bf16* __restrict__ vtb,
                                                   const float* __restrict__ decay_param,
                                                   __bf16* __restrict__ outs,
                                                   float* __restrict__ Sout) {
    __shared__ __align__(32) __bf16 Qa [CC * MM];   // A-layout  (t, k=m)
    __shared__ __align__(32) __bf16 KtB[CC * MM];   // B-layout  (n=s, k=m)
    __shared__ __align__(32) __bf16 KtA[CC * MM];   // A-layout  (m, k=s)
    __shared__ __align__(32) __bf16 Va [CC * MM];   // B-layout  (n=e, k=s)
    __shared__ __align__(32) __bf16 Ps [CC * CC];   // A-layout  (t, k=s)
    __shared__ __align__(32) __bf16 Sb [MM * MM];   // B-layout  (n=e, k=m)
    __shared__ float dp[CC + 1];

    const int b = blockIdx.x, tid = threadIdx.x;
    const int lane = tid & 31, wave = tid >> 5;
    const int half = lane >> 4, n15 = lane & 15;

    float decay = DECAY_MIN + sigmoidf_(decay_param[0]) * (DECAY_MAX - DECAY_MIN);
    if (tid <= CC) dp[tid] = __powf(decay, (float)tid);
    for (int i = tid; i < MM * MM; i += 256) Sb[i] = (__bf16)0.f;

    v8f zeroV = {};
    v8f Sacc[2] = {zeroV, zeroV};
    __syncthreads();

    for (int ch = 0; ch < LL / CC; ++ch) {
        const int rbase = b * LL + ch * CC;
        // ---- stage chunk: 4 x 2048 bf16-pair async copies ------------------
        for (int p = tid; p < 2048; p += 256) {
            int r = p >> 5, c2 = (p & 31) << 1;
            cp_async_b32(qb  + (size_t)(rbase + r) * MM + c2,            &Qa [fslot(r, c2, 4)]);
            cp_async_b32(kgb + (size_t)(rbase + r) * MM + c2,            &KtB[fslot(r, c2, 4)]);
            cp_async_b32(ktb + (size_t)(b * MM + r) * LL + ch * CC + c2, &KtA[fslot(r, c2, 4)]);
            cp_async_b32(vtb + (size_t)(b * MM + r) * LL + ch * CC + c2, &Va [fslot(r, c2, 4)]);
        }
        cp_async_wait();
        __syncthreads();

        // ---- P~ = (Q Kg^T) * decay^(t-s) * [s<=t] --------------------------
#pragma unroll
        for (int tt = 0; tt < 2; ++tt) {
            int T = wave * 2 + tt, ti = T >> 2, tj = T & 3;
            v8f acc = {};
#pragma unroll
            for (int kb = 0; kb < 2; ++kb) {
                v16bf a  = frag_ld(Qa,  kb * 4 + ti, lane);
                v16bf bf = frag_ld(KtB, kb * 4 + tj, lane);
                acc = WMMA_BF16(a, bf, acc);
            }
            int s = tj * 16 + n15;
#pragma unroll
            for (int r = 0; r < 8; ++r) {
                int t = ti * 16 + r + 8 * half;
                float w = (s <= t) ? dp[t - s] : 0.f;
                Ps[fslot(t, s, 4)] = (__bf16)(acc[r] * w);
            }
        }
        __syncthreads();

        // ---- out = P~ @ Vg + (decay^(t+1) Q) @ S_prev ----------------------
#pragma unroll
        for (int tt = 0; tt < 2; ++tt) {
            int T = wave * 2 + tt, ti = T >> 2, tj = T & 3;
            v8f acc = {};
            float qs = dp[ti * 16 + n15 + 1];
#pragma unroll
            for (int kb = 0; kb < 2; ++kb) {
                v16bf a  = frag_ld(Ps, kb * 4 + ti, lane);
                v16bf bf = frag_ld(Va, kb * 4 + tj, lane);
                acc = WMMA_BF16(a, bf, acc);
                v16bf aq = frag_ld(Qa, kb * 4 + ti, lane);
                v16bf aqs;
#pragma unroll
                for (int i = 0; i < 16; ++i) aqs[i] = (__bf16)((float)aq[i] * qs);
                v16bf bs = frag_ld(Sb, kb * 4 + tj, lane);
                acc = WMMA_BF16(aqs, bs, acc);
            }
            int n = tj * 16 + n15;
#pragma unroll
            for (int r = 0; r < 8; ++r) {
                int t = ti * 16 + r + 8 * half;
                outs[(size_t)(rbase + t) * MM + n] = (__bf16)acc[r];
            }
        }
        __syncthreads();

        // ---- S = decay^C * S + Khat^T @ Vg (state in registers) ------------
        float dC = dp[CC];
#pragma unroll
        for (int tt = 0; tt < 2; ++tt) {
            int T = wave * 2 + tt, ti = T >> 2, tj = T & 3;
            v8f acc;
#pragma unroll
            for (int r = 0; r < 8; ++r) acc[r] = dC * Sacc[tt][r];
#pragma unroll
            for (int kb = 0; kb < 2; ++kb) {
                v16bf ak = frag_ld(KtA, kb * 4 + ti, lane);
                v16bf aks;
#pragma unroll
                for (int i = 0; i < 16; ++i) {
                    int s = kb * 32 + wmma_kk(i, half);
                    aks[i] = (__bf16)((float)ak[i] * dp[CC - 1 - s]);
                }
                v16bf bf = frag_ld(Va, kb * 4 + tj, lane);
                acc = WMMA_BF16(aks, bf, acc);
            }
            Sacc[tt] = acc;
            int n = tj * 16 + n15;
#pragma unroll
            for (int r = 0; r < 8; r += 2) {            // packed b32 writes
                int m = ti * 16 + r + 8 * half;
                union { unsigned int u; __bf16 h[2]; } pk;
                pk.h[0] = (__bf16)acc[r];
                pk.h[1] = (__bf16)acc[r + 1];
                *(unsigned int*)&Sb[fslot(n, m, 4)] = pk.u;
            }
        }
        __syncthreads();
    }

    // final state -> output (f32)
#pragma unroll
    for (int tt = 0; tt < 2; ++tt) {
        int T = wave * 2 + tt, ti = T >> 2, tj = T & 3;
        int n = tj * 16 + n15;
#pragma unroll
        for (int r = 0; r < 8; ++r) {
            int m = ti * 16 + r + 8 * half;
            Sout[(size_t)b * MM * MM + m * MM + n] = Sacc[tt][r];
        }
    }
}

// ---------------------------------------------------------------------------
// Kernel 4: y = outs @ Wo + bo with LDS tiling. Block = 128 rows x 128 cols.
__global__ __launch_bounds__(256) void out_kernel(const __bf16* __restrict__ outs,
                                                  const float* __restrict__ Wo,
                                                  const float* __restrict__ bo,
                                                  float* __restrict__ y) {
    __shared__ __align__(32) __bf16 Oa[128 * 64];   // A-layout, 16 frags
    __shared__ __align__(32) __bf16 Wb[64 * 128];   // B-layout, 16 frags

    const int tid = threadIdx.x, lane = tid & 31, wave = tid >> 5;
    const int br = blockIdx.x >> 3, bc = blockIdx.x & 7;
    const int row0 = br * 128, col0 = bc * 128;

    for (int p = tid; p < 4096; p += 256) {           // outs: pure copy -> async
        int r = p >> 5, m2 = (p & 31) << 1;
        cp_async_b32(outs + (size_t)(row0 + r) * MM + m2, &Oa[fslot(r, m2, 8)]);
    }
    for (int p = tid; p < 4096; p += 256) {           // Wo: f32 -> bf16 pairs
        int c = p >> 5, k2 = (p & 31) << 1;
        float w0 = Wo[(size_t)k2 * DD + col0 + c];
        float w1 = Wo[(size_t)(k2 + 1) * DD + col0 + c];
        union { unsigned int u; __bf16 h[2]; } pk;
        pk.h[0] = (__bf16)w0; pk.h[1] = (__bf16)w1;
        *(unsigned int*)&Wb[fslot(c, k2, 8)] = pk.u;
    }
    cp_async_wait();
    __syncthreads();

    const int half = lane >> 4, n15 = lane & 15;
    v16bf a0 = frag_ld(Oa, 0 * 8 + wave, lane);
    v16bf a1 = frag_ld(Oa, 1 * 8 + wave, lane);
#pragma unroll
    for (int cj = 0; cj < 8; ++cj) {
        v8f acc = {};
        v16bf b0 = frag_ld(Wb, 0 * 8 + cj, lane);
        acc = WMMA_BF16(a0, b0, acc);
        v16bf b1 = frag_ld(Wb, 1 * 8 + cj, lane);
        acc = WMMA_BF16(a1, b1, acc);
        int n = col0 + cj * 16 + n15;
        float bias = bo[n];
#pragma unroll
        for (int r = 0; r < 8; ++r) {
            int m = row0 + wave * 16 + r + 8 * half;
            y[(size_t)m * DD + n] = acc[r] + bias;
        }
    }
}

// ---------------------------------------------------------------------------
extern "C" void kernel_launch(void* const* d_in, const int* in_sizes, int n_in,
                              void* d_out, int out_size, void* d_ws, size_t ws_size,
                              hipStream_t stream) {
    const float* x           = (const float*)d_in[0];
    const float* Wq          = (const float*)d_in[1];
    const float* Wk          = (const float*)d_in[2];
    const float* Wv          = (const float*)d_in[3];
    const float* Wo          = (const float*)d_in[4];
    const float* bo          = (const float*)d_in[5];
    const float* Wg          = (const float*)d_in[6];
    const float* bg          = (const float*)d_in[7];
    const float* decay_param = (const float*)d_in[8];

    float* y    = (float*)d_out;                    // (B,L,D) f32
    float* Sout = y + (size_t)BB * LL * DD;         // (B,M,M) f32

    // workspace: 5 bf16 buffers (q, g*k, (g*k)^T, (g*v)^T, outs) + f32 gates
    char* ws = (char*)d_ws;
    size_t nQKV = (size_t)NROWS * MM;
    __bf16* qb  = (__bf16*)ws;  ws += nQKV * sizeof(__bf16);
    __bf16* kgb = (__bf16*)ws;  ws += nQKV * sizeof(__bf16);
    __bf16* ktb = (__bf16*)ws;  ws += nQKV * sizeof(__bf16);
    __bf16* vtb = (__bf16*)ws;  ws += nQKV * sizeof(__bf16);
    __bf16* ob  = (__bf16*)ws;  ws += nQKV * sizeof(__bf16);
    float*  gates = (float*)ws;

    gate_kernel<<<NROWS / 256, 256, 0, stream>>>(x, Wg, bg, gates);
    proj_kernel<<<NROWS / 128, 256, 0, stream>>>(x, Wq, Wk, Wv, gates, qb, kgb, ktb, vtb);
    scan_kernel<<<BB, 256, 0, stream>>>(qb, kgb, ktb, vtb, decay_param, ob, Sout);
    out_kernel<<<(NROWS / 128) * (DD / 128), 256, 0, stream>>>(ob, Wo, bo, y);
}
